// GraphConvolution_50646254354782
// MI455X (gfx1250) — matrix operand order repaired and checked by reading, not compiled, not run on hardware
//
#include <hip/hip_runtime.h>
#include <stdint.h>

#define N_NODES 100000
#define N_EDGES 1600000
#define D_IN    256
#define D_OUT   128

typedef __attribute__((ext_vector_type(16))) __bf16          v16bf;
typedef __attribute__((ext_vector_type(16))) unsigned short  v16u;
typedef __attribute__((ext_vector_type(8)))  unsigned short  v8u;
typedef __attribute__((ext_vector_type(4)))  unsigned short  v4u;
typedef __attribute__((ext_vector_type(8)))  float           v8f;

// fp32 -> bf16 with round-to-nearest-even
__device__ __forceinline__ unsigned short f32_to_bf16(float f) {
  unsigned int u = __float_as_uint(f);
  u += 0x7FFFu + ((u >> 16) & 1u);
  return (unsigned short)(u >> 16);
}

// -----------------------------------------------------------------------------
// Kernel 1: support[N,128] = in_feature[N,256] @ weight[256,128]
// bf16 WMMA (v_wmma_f32_16x16x32_bf16), fp32 accumulate.
// Block = 256 threads (8 waves), 128 node rows per block.
// LDS: weight bf16 [256x128] (64KB) + X tile bf16 [128x256] (64KB) = 128KB.
// Wave w computes rows [16w,16w+16) x all 128 output cols (8 16x16 tiles).
// -----------------------------------------------------------------------------
__global__ void gcn_gemm_bf16_wmma(const float* __restrict__ A,   // [N_NODES,256]
                                   const float* __restrict__ W,   // [256,128]
                                   float* __restrict__ S)         // [N_NODES,128]
{
  extern __shared__ unsigned short smem[];
  unsigned short* lds_w = smem;                  // W[k][n] at k*128 + n
  unsigned short* lds_a = smem + 256 * 128;      // A[r][k] at r*256 + k

  const int tid  = threadIdx.x;
  const int base = blockIdx.x * 128;

  // --- cooperative load: weight fp32 -> bf16 into LDS (coalesced float4) ---
  {
    const float4* W4 = (const float4*)W;
    #pragma unroll 4
    for (int i = tid; i < (256 * 128) / 4; i += 256) {
      float4 w = W4[i];
      v4u p;
      p[0] = f32_to_bf16(w.x); p[1] = f32_to_bf16(w.y);
      p[2] = f32_to_bf16(w.z); p[3] = f32_to_bf16(w.w);
      *(v4u*)(lds_w + i * 4) = p;
    }
  }
  // --- cooperative load: X tile fp32 -> bf16 into LDS (row-clamped) ---
  {
    const float4* A4 = (const float4*)A;
    #pragma unroll 4
    for (int i = tid; i < (128 * 256) / 4; i += 256) {
      int row  = i >> 6;           // 64 float4 per 256-wide row
      int col4 = i & 63;
      int gr   = base + row;
      if (gr >= N_NODES) gr = 0;   // clamp: garbage rows never stored
      float4 a = A4[gr * 64 + col4];
      v4u p;
      p[0] = f32_to_bf16(a.x); p[1] = f32_to_bf16(a.y);
      p[2] = f32_to_bf16(a.z); p[3] = f32_to_bf16(a.w);
      *(v4u*)(lds_a + i * 4) = p;
    }
  }
  __syncthreads();

  const int wv   = tid >> 5;   // wave 0..7 -> row strip
  const int lane = tid & 31;
  const int m    = lane & 15;  // A: M index / D: N index
  const int h    = lane >> 4;  // half-wave select

  v8f acc[8];
  const v8f vzero = {0.f, 0.f, 0.f, 0.f, 0.f, 0.f, 0.f, 0.f};
  #pragma unroll
  for (int t = 0; t < 8; ++t) acc[t] = vzero;

  const unsigned short* arow = lds_a + (wv * 16 + m) * 256;

  #pragma unroll
  for (int s = 0; s < 8; ++s) {            // K steps of 32 over D_IN=256
    const int kb = 32 * s;
    // A fragment 16x32 bf16: lane half h selects K offset +8; VGPR group +16.
    // elems[0..7]  = A[m][kb + h*8 .. +7], elems[8..15] = A[m][kb+16+h*8 .. +7]
    v8u a_lo = *(const v8u*)(arow + kb + h * 8);
    v8u a_hi = *(const v8u*)(arow + kb + 16 + h * 8);
    v16u au  = __builtin_shufflevector(a_lo, a_hi,
                 0, 1, 2, 3, 4, 5, 6, 7, 8, 9, 10, 11, 12, 13, 14, 15);
    v16bf af = __builtin_bit_cast(v16bf, au);

    // B fragments 32x16 bf16: lane = N col (16t+m), elems = K in same order as A.
    #pragma unroll
    for (int t = 0; t < 8; ++t) {
      v16u bu;
      #pragma unroll
      for (int e = 0; e < 16; ++e) {
        const int k = kb + (e < 8 ? h * 8 + e : 16 + h * 8 + (e - 8));
        bu[e] = lds_w[k * 128 + 16 * t + m];
      }
      v16bf bf = __builtin_bit_cast(v16bf, bu);
      acc[t] = __builtin_amdgcn_wmma_f32_16x16x32_bf16(
                   /*neg_a=*/false, af, /*neg_b=*/false, bf,
                   /*c_mod=*/(short)0, acc[t],
                   /*reuse_a=*/false, /*reuse_b=*/false);
    }
  }

  // D layout: lanes 0-15 N=lane M=v; lanes 16-31 N=lane-16 M=v+8.
  #pragma unroll
  for (int t = 0; t < 8; ++t) {
    const int col = 16 * t + m;
    #pragma unroll
    for (int v = 0; v < 8; ++v) {
      int gr = base + wv * 16 + v + 8 * h;
      if (gr < N_NODES) S[gr * 128 + col] = acc[t][v];
    }
  }
}

// -----------------------------------------------------------------------------
// Kernel 2: out[n][d] = bias[d]  (atomics accumulate edge messages on top)
// -----------------------------------------------------------------------------
__global__ void gcn_init_out(const float* __restrict__ bias, float* __restrict__ out)
{
  int i = blockIdx.x * blockDim.x + threadIdx.x;   // over N_NODES*32 float4
  if (i < N_NODES * 32) {
    ((float4*)out)[i] = ((const float4*)bias)[i & 31];
  }
}

// -----------------------------------------------------------------------------
// Kernel 3: edge-parallel scatter. One wave per edge; lane holds one float4
// (32 lanes x 4 = 128 features). support (51MB) is L2-resident on MI455X.
// -----------------------------------------------------------------------------
__global__ void gcn_spmm_scatter(const int*   __restrict__ rows,
                                 const int*   __restrict__ cols,
                                 const float* __restrict__ vals,
                                 const float* __restrict__ S,
                                 float* __restrict__ out)
{
  const int e    = (blockIdx.x * blockDim.x + threadIdx.x) >> 5;
  const int lane = threadIdx.x & 31;
  if (e >= N_EDGES) return;

  const int   r = rows[e];
  const int   c = cols[e];
  const float v = vals[e];

  float4 s = ((const float4*)S)[c * 32 + lane];
  float* o = out + r * 128 + lane * 4;
  unsafeAtomicAdd(o + 0, v * s.x);
  unsafeAtomicAdd(o + 1, v * s.y);
  unsafeAtomicAdd(o + 2, v * s.z);
  unsafeAtomicAdd(o + 3, v * s.w);
}

// -----------------------------------------------------------------------------
extern "C" void kernel_launch(void* const* d_in, const int* in_sizes, int n_in,
                              void* d_out, int out_size, void* d_ws, size_t ws_size,
                              hipStream_t stream)
{
  const float* in_feature = (const float*)d_in[0];
  const int*   edge_rows  = (const int*)  d_in[1];
  const int*   edge_cols  = (const int*)  d_in[2];
  const float* edge_vals  = (const float*)d_in[3];
  const float* weight     = (const float*)d_in[4];
  const float* bias       = (const float*)d_in[5];
  float*       out        = (float*)d_out;
  float*       support    = (float*)d_ws;        // N_NODES*128 fp32 = 51.2 MB

  // 1) dense transform via bf16 WMMA
  const int    gemm_blocks = (N_NODES + 127) / 128;          // 782
  const size_t lds_bytes   = (size_t)(256 * 128 + 128 * 256) * sizeof(unsigned short); // 128KB
  gcn_gemm_bf16_wmma<<<gemm_blocks, 256, lds_bytes, stream>>>(in_feature, weight, support);

  // 2) bias init of output
  gcn_init_out<<<(N_NODES * 32 + 255) / 256, 256, 0, stream>>>(bias, out);

  // 3) COO SpMM scatter (8 edges per 256-thread block)
  gcn_spmm_scatter<<<(N_EDGES + 7) / 8, 256, 0, stream>>>(edge_rows, edge_cols,
                                                          edge_vals, support, out);
}
